// DeepFM_67534065762719
// MI455X (gfx1250) — compile-verified, matrix operand order checked by reading, not run on hardware
//
#include <hip/hip_runtime.h>
#include <hip/hip_bf16.h>
#include <math.h>

// ---------------- problem constants ----------------
#define B_     16384
#define F_     26
#define V_     100000
#define E_     16
#define D_     13
#define DNNIN_ 429
#define K1_    448      // DNN_IN padded to multiple of 32 (WMMA K)
#define N1_    256
#define N2_    128

typedef __attribute__((ext_vector_type(16))) __bf16 v16bf;
typedef __attribute__((ext_vector_type(8)))  __bf16 v8bf;
typedef __attribute__((ext_vector_type(8)))  float  v8f;

// ---------------- workspace layout (bytes) ----------------
// Hbf   : [B, 448] bf16      @ 0          (14,680,064)
// A1bf  : [B, 256] bf16      @ 14,680,064 ( 8,388,608)
// W1bf  : [256,448] bf16     @ 23,068,672 (   229,376)
// W2bf  : [128,256] bf16     @ 23,298,048 (    65,536)
// fmlog : [B] f32            @ 23,363,584 (    65,536)
#define OFF_A1   14680064u
#define OFF_W1   23068672u
#define OFF_W2   23298048u
#define OFF_FM   23363584u

// ============================================================
// Kernel 1: convert W1 (256x429, pad K to 448) and W2 (128x256)
// to bf16 in workspace.
// ============================================================
__global__ void prep_weights(const float* __restrict__ W1,
                             const float* __restrict__ W2,
                             __bf16* __restrict__ W1bf,
                             __bf16* __restrict__ W2bf) {
    const int t = blockIdx.x * blockDim.x + threadIdx.x;
    const int n1 = N1_ * K1_;                 // 114688
    if (t < n1) {
        const int n = t / K1_;
        const int k = t - n * K1_;
        const float v = (k < DNNIN_) ? W1[n * DNNIN_ + k] : 0.0f;
        W1bf[t] = (__bf16)v;
    } else {
        const int u = t - n1;
        if (u < N2_ * N1_) W2bf[u] = (__bf16)W2[u];
    }
}

// ============================================================
// Kernel 2: embedding gather + FM 1st/2nd order + build bf16 H.
// One wave32 per batch row (4 rows / 128-thread block).
//   lane f < 26 : gathers field f's 16-wide FM embedding + linear term
//   lanes 0..15 : per-component column sums for the cross term
// ============================================================
__global__ void gather_fm(const int*   __restrict__ Xs,
                          const float* __restrict__ Xd,
                          const float* __restrict__ linw,
                          const float* __restrict__ fmw,
                          const float* __restrict__ dw,
                          const float* __restrict__ db,
                          __bf16* __restrict__ Hbf,
                          float*  __restrict__ fmlog) {
    __shared__ __align__(16) float fm_s[4][F_][E_];
    const int lane = threadIdx.x & 31;
    const int wave = threadIdx.x >> 5;
    const int b    = blockIdx.x * 4 + wave;

    float lin = 0.0f, s2 = 0.0f;
    if (lane < F_) {
        const int idx = Xs[b * F_ + lane];
        const size_t base = (size_t)lane * V_ + (size_t)idx;
        lin = linw[base];
        const float4* p = (const float4*)(fmw + base * E_);
        const float4 v0 = p[0], v1 = p[1], v2 = p[2], v3 = p[3];

        float4* fs = (float4*)&fm_s[wave][lane][0];
        fs[0] = v0; fs[1] = v1; fs[2] = v2; fs[3] = v3;

        v8bf h0, h1;
        h0[0] = (__bf16)v0.x; h0[1] = (__bf16)v0.y; h0[2] = (__bf16)v0.z; h0[3] = (__bf16)v0.w;
        h0[4] = (__bf16)v1.x; h0[5] = (__bf16)v1.y; h0[6] = (__bf16)v1.z; h0[7] = (__bf16)v1.w;
        h1[0] = (__bf16)v2.x; h1[1] = (__bf16)v2.y; h1[2] = (__bf16)v2.z; h1[3] = (__bf16)v2.w;
        h1[4] = (__bf16)v3.x; h1[5] = (__bf16)v3.y; h1[6] = (__bf16)v3.z; h1[7] = (__bf16)v3.w;
        v8bf* hd = (v8bf*)(Hbf + (size_t)b * K1_ + lane * E_);
        hd[0] = h0; hd[1] = h1;

        s2  = v0.x*v0.x + v0.y*v0.y + v0.z*v0.z + v0.w*v0.w;
        s2 += v1.x*v1.x + v1.y*v1.y + v1.z*v1.z + v1.w*v1.w;
        s2 += v2.x*v2.x + v2.y*v2.y + v2.z*v2.z + v2.w*v2.w;
        s2 += v3.x*v3.x + v3.y*v3.y + v3.z*v3.z + v3.w*v3.w;
    }

    // dense features: contribute to linear part + tail of H row
    float dp = 0.0f;
    if (lane < D_) {
        const float xv = Xd[b * D_ + lane];
        dp = xv * dw[lane];
        Hbf[(size_t)b * K1_ + F_ * E_ + lane] = (__bf16)xv;
    }
    // zero-pad H[429..447]
    if (lane < (K1_ - DNNIN_))
        Hbf[(size_t)b * K1_ + DNNIN_ + lane] = (__bf16)0.0f;

    // wave reduce: linear part and sum-of-squares
    float lp  = lin + dp;
    float s2t = s2;
    #pragma unroll
    for (int m = 16; m >= 1; m >>= 1) {
        lp  += __shfl_xor(lp,  m, 32);
        s2t += __shfl_xor(s2t, m, 32);
    }

    __syncthreads();   // fm_s visibility

    // square-of-sum: lane e sums column e over 26 fields
    float cr = 0.0f;
    if (lane < E_) {
        float se = 0.0f;
        #pragma unroll
        for (int f = 0; f < F_; ++f) se += fm_s[wave][f][lane];
        cr = se * se;
    }
    #pragma unroll
    for (int m = 16; m >= 1; m >>= 1) cr += __shfl_xor(cr, m, 32);

    if (lane == 0)
        fmlog[b] = lp + db[0] + 0.5f * (cr - s2t);
}

// ============================================================
// WMMA fragment loader helper: bf16 A/B fragment per ISA layout.
// lanes 0-15 : row/col (lane&15), K in {k0..k0+7, k0+16..k0+23}
// lanes 16-31: row/col (lane&15), K in {k0+8..k0+15, k0+24..k0+31}
// ============================================================
__device__ __forceinline__ v16bf load_frag(const __bf16* p) {
    const v8bf lo = *(const v8bf*)(p);
    const v8bf hi = *(const v8bf*)(p + 16);
    v16bf r;
    #pragma unroll
    for (int i = 0; i < 8; ++i) { r[i] = lo[i]; r[i + 8] = hi[i]; }
    return r;
}

// ============================================================
// Kernel 3: A1 = relu(H @ W1^T + b1) in bf16.
// M=16384, N=256, K=448. 8 waves/block, 1 tile (16x16) per wave.
// block covers 16 rows x 128 cols; grid = (1024, 2).
// The compiler software-pipelines the b128 fragment loads ~6 deep
// past each WMMA (seen in disasm) — no explicit prefetch needed.
// ============================================================
__global__ void gemm1(const __bf16* __restrict__ Hbf,
                      const __bf16* __restrict__ W1bf,
                      const float*  __restrict__ b1,
                      __bf16* __restrict__ A1) {
    const int lane = threadIdx.x & 31;
    const int wave = threadIdx.x >> 5;
    const int m0   = blockIdx.x * 16;
    const int n0   = blockIdx.y * 128 + wave * 16;
    const int haf  = (lane >> 4) * 8;
    const int rowA = m0 + (lane & 15);
    const int col  = n0 + (lane & 15);

    const __bf16* ap = Hbf  + (size_t)rowA * K1_ + haf;
    const __bf16* bp = W1bf + (size_t)col  * K1_ + haf;

    v8f c = {};
    #pragma unroll
    for (int kk = 0; kk < K1_ / 32; ++kk) {
        const v16bf a = load_frag(ap);
        const v16bf b = load_frag(bp);
        c = __builtin_amdgcn_wmma_f32_16x16x32_bf16(false, a, false, b,
                                                    (short)0, c, false, false);
        ap += 32; bp += 32;
    }

    const float bias = b1[col];
    #pragma unroll
    for (int r = 0; r < 8; ++r) {
        const int row = m0 + r + haf;
        const float v = c[r] + bias;
        A1[(size_t)row * N1_ + col] = (__bf16)(v > 0.0f ? v : 0.0f);
    }
}

// ============================================================
// Kernel 4: h2 = relu(A1 @ W2^T + b2); logit = h2 . Wout;
//           out = sigmoid(fm_logit + logit). Fully fused.
// M=16384, N=128, K=256. 8 waves/block cover all 128 cols of 16 rows.
// ============================================================
__global__ void gemm2_out(const __bf16* __restrict__ A1,
                          const __bf16* __restrict__ W2bf,
                          const float*  __restrict__ b2,
                          const float*  __restrict__ Wout,
                          const float*  __restrict__ fmlog,
                          float* __restrict__ out) {
    __shared__ float acc[16];
    const int lane = threadIdx.x & 31;
    const int wave = threadIdx.x >> 5;
    const int m0   = blockIdx.x * 16;
    const int n0   = wave * 16;
    const int haf  = (lane >> 4) * 8;
    const int rowA = m0 + (lane & 15);
    const int col  = n0 + (lane & 15);

    if (threadIdx.x < 16) acc[threadIdx.x] = 0.0f;
    __syncthreads();

    const __bf16* ap = A1   + (size_t)rowA * N1_ + haf;
    const __bf16* bp = W2bf + (size_t)col  * N1_ + haf;

    v8f c = {};
    #pragma unroll
    for (int kk = 0; kk < N1_ / 32; ++kk) {
        const v16bf a = load_frag(ap);
        const v16bf b = load_frag(bp);
        c = __builtin_amdgcn_wmma_f32_16x16x32_bf16(false, a, false, b,
                                                    (short)0, c, false, false);
        ap += 32; bp += 32;
    }

    const float bias = b2[col];
    const float wo   = Wout[col];
    float part[8];
    #pragma unroll
    for (int r = 0; r < 8; ++r) {
        float v = c[r] + bias;
        v = v > 0.0f ? v : 0.0f;
        part[r] = v * wo;           // contribution to row (m0+r+haf)'s Wout dot
    }
    // reduce over the 16 lanes of each half (cols n0..n0+15)
    #pragma unroll
    for (int m = 8; m >= 1; m >>= 1) {
        #pragma unroll
        for (int r = 0; r < 8; ++r) part[r] += __shfl_xor(part[r], m, 32);
    }
    if ((lane & 15) == 0) {
        #pragma unroll
        for (int r = 0; r < 8; ++r) atomicAdd(&acc[r + haf], part[r]);
    }
    __syncthreads();

    if (threadIdx.x < 16) {
        const int row = m0 + threadIdx.x;
        const float x = fmlog[row] + acc[threadIdx.x];
        out[row] = 1.0f / (1.0f + __expf(-x));
    }
}

// ============================================================
// Host-side launcher
// ============================================================
extern "C" void kernel_launch(void* const* d_in, const int* in_sizes, int n_in,
                              void* d_out, int out_size, void* d_ws, size_t ws_size,
                              hipStream_t stream) {
    const int*   Xs   = (const int*)  d_in[0];   // X_sparse [B,F]
    const float* Xd   = (const float*)d_in[1];   // X_dense  [B,13]
    const float* linw = (const float*)d_in[2];   // lin_emb_w [F,V,1]
    const float* fmw  = (const float*)d_in[3];   // fm_emb_w  [F,V,E]
    const float* dw   = (const float*)d_in[4];   // dense_w [1,13]
    const float* db   = (const float*)d_in[5];   // dense_b [1]
    const float* W1   = (const float*)d_in[6];   // [256,429]
    const float* b1   = (const float*)d_in[7];   // [256]
    const float* W2   = (const float*)d_in[8];   // [128,256]
    const float* b2   = (const float*)d_in[9];   // [128]
    const float* Wout = (const float*)d_in[10];  // [1,128]
    float* out = (float*)d_out;

    char* ws = (char*)d_ws;
    __bf16* Hbf   = (__bf16*)(ws);
    __bf16* A1    = (__bf16*)(ws + OFF_A1);
    __bf16* W1bf  = (__bf16*)(ws + OFF_W1);
    __bf16* W2bf  = (__bf16*)(ws + OFF_W2);
    float*  fmlog = (float*) (ws + OFF_FM);

    const int prep_elems = N1_ * K1_ + N2_ * N1_;
    prep_weights<<<(prep_elems + 255) / 256, 256, 0, stream>>>(W1, W2, W1bf, W2bf);
    gather_fm<<<B_ / 4, 128, 0, stream>>>(Xs, Xd, linw, fmw, dw, db, Hbf, fmlog);
    gemm1<<<dim3(B_ / 16, 2), 256, 0, stream>>>(Hbf, W1bf, b1, A1);
    gemm2_out<<<B_ / 16, 256, 0, stream>>>(A1, W2bf, b2, Wout, fmlog, out);
}
